// CognitiveTransformerLayer_54503134986910
// MI455X (gfx1250) — matrix operand order repaired and checked by reading, compiled, not run on hardware
//
#include <hip/hip_runtime.h>
#include <hip/hip_bf16.h>
#include <cstdint>

typedef __bf16 bf16;
typedef __attribute__((ext_vector_type(16))) __bf16 v16bf;
typedef __attribute__((ext_vector_type(8)))  float  v8f;

#define WMMA_BF16(a, b, c) \
  __builtin_amdgcn_wmma_f32_16x16x32_bf16(false, (a), false, (b), (short)0, (c), false, false)

// ---------------------------------------------------------------- constants
static constexpr int   Bz   = 2;
static constexpr int   S    = 2048;
static constexpr int   Hd   = 2048;      // hidden
static constexpr int   NH   = 16;
static constexpr int   NKV  = 8;
static constexpr int   HD   = 128;       // head dim
static constexpr int   FF   = 5632;
static constexpr int   MB   = Bz * S;    // 4096 rows
static constexpr float EPS  = 1e-6f;
static constexpr float SCAL = 0.08838834764831845f;  // 1/sqrt(128)

// ---------------------------------------------------------------- CDNA5 async helpers
// LDS byte offset = low 32 bits of the flat pointer (LDS aperture keeps offset
// in addr[31:0], ISA §10.2).
__device__ __forceinline__ uint32_t lds_off(const void* p) {
  return (uint32_t)(uintptr_t)p;
}
// global -> LDS async copy, 16 bytes per lane, tracked by ASYNCcnt.
__device__ __forceinline__ void async_copy_b128(uint32_t lds, const void* gaddr) {
  asm volatile("global_load_async_to_lds_b128 %0, %1, off"
               :: "v"(lds), "v"(gaddr) : "memory");
}
#define WAIT_ASYNC(n)  asm volatile("s_wait_asynccnt " #n ::: "memory")
#define WAIT_DS0()     asm volatile("s_wait_dscnt 0x0" ::: "memory")

// fragment load union (two b128 chunks)
union FragU {
  uint4 q[2];
  v16bf v;
};

// ---------------------------------------------------------------- fp32 -> bf16
__global__ __launch_bounds__(256) void k_cvt_bf16(const float* __restrict__ in,
                                                  bf16* __restrict__ out, size_t n) {
  size_t i = (size_t)blockIdx.x * blockDim.x + threadIdx.x;
  size_t stride = (size_t)gridDim.x * blockDim.x;
  for (; i < n; i += stride) out[i] = (bf16)in[i];
}

// ---------------------------------------------------------------- RMSNorm -> bf16
__global__ __launch_bounds__(256) void k_rmsnorm(const float* __restrict__ x,
                                                 const float* __restrict__ w,
                                                 bf16* __restrict__ out) {
  const int row = blockIdx.x;
  const float* xr = x + (size_t)row * Hd;
  float s = 0.f;
  for (int i = threadIdx.x; i < Hd; i += 256) { float v = xr[i]; s += v * v; }
  for (int d = 1; d < 32; d <<= 1) s += __shfl_xor(s, d);
  __shared__ float red[8];
  __shared__ float invs;
  if ((threadIdx.x & 31) == 0) red[threadIdx.x >> 5] = s;
  __syncthreads();
  if (threadIdx.x == 0) {
    float t = 0.f;
    #pragma unroll
    for (int i = 0; i < 8; ++i) t += red[i];
    invs = rsqrtf(t / (float)Hd + EPS);
  }
  __syncthreads();
  const float r = invs;
  for (int i = threadIdx.x; i < Hd; i += 256)
    out[(size_t)row * Hd + i] = (bf16)(xr[i] * r * w[i]);
}

// ---------------------------------------------------------------- SwiGLU (bf16 in/out)
__global__ __launch_bounds__(256) void k_swiglu(const bf16* __restrict__ g,
                                                const bf16* __restrict__ u,
                                                bf16* __restrict__ out, size_t n) {
  size_t i = (size_t)blockIdx.x * blockDim.x + threadIdx.x;
  size_t stride = (size_t)gridDim.x * blockDim.x;
  for (; i < n; i += stride) {
    float gv = (float)g[i];
    float sv = gv / (1.f + __expf(-gv));
    out[i] = (bf16)(sv * (float)u[i]);
  }
}

// ---------------------------------------------------------------- GEMM: C = A @ W^T (+resid)
// A: [M,K] bf16 row-major, W: [N,K] bf16 row-major. M mult of 128, N mult of 256,
// K mult of 32. Block tile 128x256, 8 waves, 64x64 per wave (16 WMMAs / k-step).
// Double-buffered LDS staged with global_load_async_to_lds_b128.
enum { EPI_BF16 = 0, EPI_F32 = 1, EPI_F32_RESID = 2 };

#define GT_M 128
#define GT_N 256
#define GLDS 40   // LDS row stride in bf16 elems (32 + 8 pad; 80B rows, 16B aligned)

template <int EPI>
__global__ __launch_bounds__(256)
void k_gemm(const bf16* __restrict__ A, const bf16* __restrict__ W,
            float* __restrict__ Cf, bf16* __restrict__ Cb,
            const float* __restrict__ resid, int M, int N, int K) {
  __shared__ __align__(16) bf16 As[2][GT_M * GLDS];
  __shared__ __align__(16) bf16 Bs[2][GT_N * GLDS];

  const int tid  = threadIdx.x;
  const int lane = tid & 31;
  const int wave = tid >> 5;          // 0..7
  const int wm   = wave >> 2;         // 0..1  (64 rows)
  const int wn   = wave & 3;          // 0..3  (64 cols)
  const int m0   = blockIdx.y * GT_M;
  const int n0   = blockIdx.x * GT_N;

  const int lr = tid >> 1;            // 0..127
  const int lc = (tid & 1) * 16;      // 0 or 16

  const int nrow  = lane & 15;
  const int half  = lane >> 4;
  const int koffA = half ? 16 : 0;    // A-frag contiguous 16 start
  const int koffB = half ? 16 : 0;    // B-frag contiguous 16 start

  v8f acc[4][4];
  #pragma unroll
  for (int i = 0; i < 4; ++i)
    #pragma unroll
    for (int j = 0; j < 4; ++j)
      acc[i][j] = (v8f){0.f, 0.f, 0.f, 0.f, 0.f, 0.f, 0.f, 0.f};

  // stage one 32-wide K panel into buffer `buf` (6 async b128 per thread)
  auto stage = [&](int buf, int k0) {
    const bf16* ga  = A + (size_t)(m0 + lr) * K + k0 + lc;
    async_copy_b128(lds_off(&As[buf][lr * GLDS + lc]),     ga);
    async_copy_b128(lds_off(&As[buf][lr * GLDS + lc + 8]), ga + 8);
    const bf16* gw0 = W + (size_t)(n0 + lr) * K + k0 + lc;
    const bf16* gw1 = W + (size_t)(n0 + 128 + lr) * K + k0 + lc;
    async_copy_b128(lds_off(&Bs[buf][lr * GLDS + lc]),            gw0);
    async_copy_b128(lds_off(&Bs[buf][lr * GLDS + lc + 8]),        gw0 + 8);
    async_copy_b128(lds_off(&Bs[buf][(128 + lr) * GLDS + lc]),    gw1);
    async_copy_b128(lds_off(&Bs[buf][(128 + lr) * GLDS + lc + 8]), gw1 + 8);
  };

  stage(0, 0);
  const int nsteps = K >> 5;
  for (int s = 0; s < nsteps; ++s) {
    const int cur = s & 1;
    __syncthreads();                      // everyone done reading buf cur^1
    if (s + 1 < nsteps) {
      stage(cur ^ 1, (s + 1) << 5);       // prefetch next panel
      WAIT_ASYNC(0x6);                    // first 6 (current panel) landed
    } else {
      WAIT_ASYNC(0x0);
    }
    __syncthreads();                      // all threads' panels landed

    v16bf bfr[4];
    #pragma unroll
    for (int ct = 0; ct < 4; ++ct) {
      const bf16* p = &Bs[cur][(wn * 64 + ct * 16 + nrow) * GLDS + koffB];
      FragU u;
      u.q[0] = *reinterpret_cast<const uint4*>(p);
      u.q[1] = *reinterpret_cast<const uint4*>(p + 8);
      bfr[ct] = u.v;
    }
    #pragma unroll
    for (int mt = 0; mt < 4; ++mt) {
      const bf16* p = &As[cur][(wm * 64 + mt * 16 + nrow) * GLDS + koffA];
      FragU u;
      u.q[0] = *reinterpret_cast<const uint4*>(p);
      u.q[1] = *reinterpret_cast<const uint4*>(p + 8);
      const v16bf afr = u.v;
      #pragma unroll
      for (int ct = 0; ct < 4; ++ct)
        acc[mt][ct] = WMMA_BF16(afr, bfr[ct], acc[mt][ct]);
    }
  }

  // epilogue: VGPR i of lane L holds (row = i + 8*half, col = L&15)
  #pragma unroll
  for (int mt = 0; mt < 4; ++mt) {
    const int rbase = m0 + wm * 64 + mt * 16 + half * 8;
    #pragma unroll
    for (int ct = 0; ct < 4; ++ct) {
      const int col = n0 + wn * 64 + ct * 16 + nrow;
      #pragma unroll
      for (int i = 0; i < 8; ++i) {
        const size_t idx = (size_t)(rbase + i) * N + col;
        if (EPI == EPI_BF16)      Cb[idx] = (bf16)acc[mt][ct][i];
        else if (EPI == EPI_F32)  Cf[idx] = acc[mt][ct][i];
        else                      Cf[idx] = acc[mt][ct][i] + resid[idx];
      }
    }
  }
}

// ---------------------------------------------------------------- flash attention
// One wave per (batch, head, 16-query tile). Q:[MB,2048] K/V:[MB,1024] bf16. O bf16.
__global__ __launch_bounds__(32)
void k_attn(const bf16* __restrict__ Q, const bf16* __restrict__ Kb,
            const bf16* __restrict__ Vb, bf16* __restrict__ O) {
  __shared__ __align__(16) bf16 Ps[16 * 40];    // P tile 16x32 (stride 40)
  __shared__ __align__(16) bf16 Vs[32 * 136];   // V tile 32x128 (stride 136)

  const int lane = threadIdx.x;
  const int q0   = blockIdx.x * 16;
  const int head = blockIdx.y;
  const int b    = blockIdx.z;
  const int kvh  = head >> 1;                   // GROUPS = 2
  const int nidx = lane & 15;
  const int half = lane >> 4;
  const int koff = half ? 16 : 0;

  // Q fragments: 4 chunks of 32 over HD=128  (b128 loads)
  v16bf qf[4];
  {
    const bf16* qbase = Q + ((size_t)(b * S + q0 + nidx)) * (NH * HD) + head * HD;
    #pragma unroll
    for (int kc = 0; kc < 4; ++kc) {
      const bf16* p = qbase + kc * 32 + koff;
      FragU u;
      u.q[0] = *reinterpret_cast<const uint4*>(p);
      u.q[1] = *reinterpret_cast<const uint4*>(p + 8);
      qf[kc] = u.v;
    }
  }

  v8f o[8];
  #pragma unroll
  for (int t = 0; t < 8; ++t) o[t] = (v8f){0.f, 0.f, 0.f, 0.f, 0.f, 0.f, 0.f, 0.f};
  float rmax[8], rsum[8];
  #pragma unroll
  for (int i = 0; i < 8; ++i) { rmax[i] = -3.0e38f; rsum[i] = 0.f; }

  const int ntiles = (q0 + 16 + 31) >> 5;       // causal: keys <= q0+15
  for (int kt = 0; kt < ntiles; ++kt) {
    const int kbase = kt * 32;

    // prev iteration's LDS reads fully retired before async writes can land
    WAIT_DS0();
    // async-stage V tile [32 keys x 128 dims]; overlaps with QK^T below
    #pragma unroll
    for (int i = 0; i < 16; ++i) {
      const int c  = i * 32 + lane;             // 0..511
      const int vr = c >> 4;                    // key row 0..31
      const int vc = (c & 15) * 8;              // dim 0..120
      async_copy_b128(lds_off(&Vs[vr * 136 + vc]),
                      Vb + ((size_t)(b * S + kbase + vr)) * (NKV * HD) + kvh * HD + vc);
    }

    // scores S[16 x 32] = Q @ K^T
    v8f s0 = (v8f){0.f,0.f,0.f,0.f,0.f,0.f,0.f,0.f};
    v8f s1 = s0;
    #pragma unroll
    for (int kc = 0; kc < 4; ++kc) {
      const int dimoff = kc * 32 + half * 16;   // B-frag: contiguous 16 dims per lane
      const bf16* kp0 = Kb + ((size_t)(b * S + kbase + nidx)) * (NKV * HD) + kvh * HD + dimoff;
      const bf16* kp1 = kp0 + (size_t)16 * (NKV * HD);   // keys +16
      FragU u0, u1;
      u0.q[0] = *reinterpret_cast<const uint4*>(kp0);
      u0.q[1] = *reinterpret_cast<const uint4*>(kp0 + 8);
      u1.q[0] = *reinterpret_cast<const uint4*>(kp1);
      u1.q[1] = *reinterpret_cast<const uint4*>(kp1 + 8);
      s0 = WMMA_BF16(qf[kc], u0.v, s0);
      s1 = WMMA_BF16(qf[kc], u1.v, s1);
    }

    // scale + causal mask + online softmax (rows spread over 16-lane halves)
    const int qg = q0 + half * 8;               // element i is row qg+i
    float corr[8];
    #pragma unroll
    for (int i = 0; i < 8; ++i) {
      const int row = qg + i;
      float e0 = s0[i] * SCAL; if (kbase + nidx      > row) e0 = -3.0e30f;
      float e1 = s1[i] * SCAL; if (kbase + nidx + 16 > row) e1 = -3.0e30f;
      float v = fmaxf(e0, e1);
      #pragma unroll
      for (int d = 1; d < 16; d <<= 1) v = fmaxf(v, __shfl_xor(v, d));
      const float nm = fmaxf(rmax[i], v);
      corr[i] = __expf(rmax[i] - nm);
      rmax[i] = nm;
      const float p0 = __expf(e0 - nm);
      const float p1 = __expf(e1 - nm);
      s0[i] = p0; s1[i] = p1;
      float ss = p0 + p1;
      #pragma unroll
      for (int d = 1; d < 16; d <<= 1) ss += __shfl_xor(ss, d);
      rsum[i] = rsum[i] * corr[i] + ss;
    }
    #pragma unroll
    for (int t = 0; t < 8; ++t)
      #pragma unroll
      for (int i = 0; i < 8; ++i) o[t][i] *= corr[i];

    // P (C-layout) -> LDS -> A-fragment layout
    #pragma unroll
    for (int i = 0; i < 8; ++i) {
      const int prow = half * 8 + i;
      Ps[prow * 40 + nidx]      = (bf16)s0[i];
      Ps[prow * 40 + 16 + nidx] = (bf16)s1[i];
    }
    WAIT_DS0();                                 // stores visible before re-read
    v16bf pf;
    {
      const bf16* p = &Ps[nidx * 40 + koff];
      FragU u;
      u.q[0] = *reinterpret_cast<const uint4*>(p);
      u.q[1] = *reinterpret_cast<const uint4*>(p + 8);
      pf = u.v;
    }

    WAIT_ASYNC(0x0);                            // V tile landed in LDS

    // O += P @ V  (8 column tiles of 16)
    #pragma unroll
    for (int ct = 0; ct < 8; ++ct) {
      const bf16* vp = &Vs[(half * 16) * 136 + ct * 16 + nidx];
      v16bf vf;
      #pragma unroll
      for (int i = 0; i < 16; ++i) vf[i] = vp[i * 136];
      o[ct] = WMMA_BF16(pf, vf, o[ct]);
    }
  }

  // epilogue: normalize and store bf16
  #pragma unroll
  for (int t = 0; t < 8; ++t)
    #pragma unroll
    for (int i = 0; i < 8; ++i) {
      const int row = q0 + half * 8 + i;
      const int col = head * HD + t * 16 + nidx;
      O[((size_t)(b * S + row)) * (NH * HD) + col] = (bf16)(o[t][i] / rsum[i]);
    }
}

// ---------------------------------------------------------------- launcher
extern "C" void kernel_launch(void* const* d_in, const int* in_sizes, int n_in,
                              void* d_out, int out_size, void* d_ws, size_t ws_size,
                              hipStream_t stream) {
  (void)in_sizes; (void)n_in; (void)out_size; (void)ws_size;
  const float* x    = (const float*)d_in[0];
  /* d_in[1] = attention_mask: exactly causal; applied analytically */
  const float* qw   = (const float*)d_in[2];
  const float* kw   = (const float*)d_in[3];
  const float* vw   = (const float*)d_in[4];
  const float* ow   = (const float*)d_in[5];
  const float* gw   = (const float*)d_in[6];
  const float* uw   = (const float*)d_in[7];
  const float* dw   = (const float*)d_in[8];
  const float* ln1  = (const float*)d_in[9];
  const float* ln2  = (const float*)d_in[10];
  float* out = (float*)d_out;

  char* wsb = (char*)d_ws;
  size_t off = 0;
  auto alloc = [&](size_t bytes) -> void* {
    void* p = wsb + off;
    off = (off + bytes + 255) & ~(size_t)255;
    return p;
  };

  bf16* wq = (bf16*)alloc((size_t)(NH * HD) * Hd * 2);
  bf16* wk = (bf16*)alloc((size_t)(NKV * HD) * Hd * 2);
  bf16* wv = (bf16*)alloc((size_t)(NKV * HD) * Hd * 2);
  bf16* wo = (bf16*)alloc((size_t)Hd * (NH * HD) * 2);
  bf16* wg = (bf16*)alloc((size_t)FF * Hd * 2);
  bf16* wu = (bf16*)alloc((size_t)FF * Hd * 2);
  bf16* wd = (bf16*)alloc((size_t)Hd * FF * 2);
  bf16* xn = (bf16*)alloc((size_t)MB * Hd * 2);
  bf16* qb = (bf16*)alloc((size_t)MB * (NH * HD) * 2);
  bf16* kb = (bf16*)alloc((size_t)MB * (NKV * HD) * 2);
  bf16* vb = (bf16*)alloc((size_t)MB * (NKV * HD) * 2);
  bf16* ab = (bf16*)alloc((size_t)MB * (NH * HD) * 2);
  float* h = (float*)alloc((size_t)MB * Hd * 4);
  bf16* gb = (bf16*)alloc((size_t)MB * FF * 2);
  bf16* ub = (bf16*)alloc((size_t)MB * FF * 2);
  bf16* h2 = (bf16*)alloc((size_t)MB * FF * 2);

  // 1) weights -> bf16
  k_cvt_bf16<<<1024, 256, 0, stream>>>(qw, wq, (size_t)(NH * HD) * Hd);
  k_cvt_bf16<<<1024, 256, 0, stream>>>(kw, wk, (size_t)(NKV * HD) * Hd);
  k_cvt_bf16<<<1024, 256, 0, stream>>>(vw, wv, (size_t)(NKV * HD) * Hd);
  k_cvt_bf16<<<1024, 256, 0, stream>>>(ow, wo, (size_t)Hd * (NH * HD));
  k_cvt_bf16<<<2048, 256, 0, stream>>>(gw, wg, (size_t)FF * Hd);
  k_cvt_bf16<<<2048, 256, 0, stream>>>(uw, wu, (size_t)FF * Hd);
  k_cvt_bf16<<<2048, 256, 0, stream>>>(dw, wd, (size_t)Hd * FF);

  // 2) pre-attention RMSNorm
  k_rmsnorm<<<MB, 256, 0, stream>>>(x, ln1, xn);

  // 3) Q/K/V projections (bf16 out)
  k_gemm<EPI_BF16><<<dim3((NH * HD) / GT_N, MB / GT_M), 256, 0, stream>>>(
      xn, wq, nullptr, qb, nullptr, MB, NH * HD, Hd);
  k_gemm<EPI_BF16><<<dim3((NKV * HD) / GT_N, MB / GT_M), 256, 0, stream>>>(
      xn, wk, nullptr, kb, nullptr, MB, NKV * HD, Hd);
  k_gemm<EPI_BF16><<<dim3((NKV * HD) / GT_N, MB / GT_M), 256, 0, stream>>>(
      xn, wv, nullptr, vb, nullptr, MB, NKV * HD, Hd);

  // 4) flash attention (causal, GQA)
  k_attn<<<dim3(S / 16, NH, Bz), 32, 0, stream>>>(qb, kb, vb, ab);

  // 5) output projection + residual -> h (fp32)
  k_gemm<EPI_F32_RESID><<<dim3(Hd / GT_N, MB / GT_M), 256, 0, stream>>>(
      ab, wo, h, nullptr, x, MB, Hd, NH * HD);

  // 6) pre-MLP RMSNorm
  k_rmsnorm<<<MB, 256, 0, stream>>>(h, ln2, xn);

  // 7) gate / up projections (bf16 out)
  k_gemm<EPI_BF16><<<dim3(FF / GT_N, MB / GT_M), 256, 0, stream>>>(
      xn, wg, nullptr, gb, nullptr, MB, FF, Hd);
  k_gemm<EPI_BF16><<<dim3(FF / GT_N, MB / GT_M), 256, 0, stream>>>(
      xn, wu, nullptr, ub, nullptr, MB, FF, Hd);

  // 8) SwiGLU
  k_swiglu<<<2048, 256, 0, stream>>>(gb, ub, h2, (size_t)MB * FF);

  // 9) down projection + residual -> out (fp32)
  k_gemm<EPI_F32_RESID><<<dim3(Hd / GT_N, MB / GT_M), 256, 0, stream>>>(
      h2, wd, out, nullptr, h, MB, Hd, FF);
}
